// PosAddRQSpline_36618891165837
// MI455X (gfx1250) — matrix operand order repaired and checked by reading, not compiled, not run on hardware
//
#include <hip/hip_runtime.h>
#include <hip/hip_bf16.h>
#include <math.h>

typedef __attribute__((ext_vector_type(16))) _Float16 v16h;
typedef __attribute__((ext_vector_type(8)))  _Float16 v8h;
typedef __attribute__((ext_vector_type(8)))  float    v8f;

#define NF 16
#define NL 8
#define HID 256
#define NB 8
#define PPF 25            // 3*NB+1 params per feature
#define PTOT 400          // NF*PPF
#define NT_WF 26          // 400 padded to 416 -> 26 tiles of 16
#define ROWS 32           // batch rows per block (2 row-tiles of 16)
#define HA_STRIDE 264     // f16 activation row stride (256 + 8 pad)
#define T_STRIDE 416      // f32 temp row stride (max(HID, padded P))

// ---------------------------------------------------------------------------
// Weight packing: f32 -> f16 in the CDNA5 B-matrix wave32 fragment layout for
// V_WMMA_F32_16X16X32_F16. Per (ktile, ntile) block of 32 lanes x 16 halves:
//   N = ntile*16 + (lane & 15)
//   K = ktile*32 + (half >> 3)*16 + (lane >> 4)*8 + (half & 7)      (half = 0..15)
// Block flat order (Wr):  (((l*4 + j)*16 + kt)*16 + nt)
// Block flat order (Wf):  ((l*16 + kt)*26 + nt)
// ---------------------------------------------------------------------------
__global__ void pack_wr(const float* __restrict__ Wr, _Float16* __restrict__ dst) {
  long i = (long)blockIdx.x * blockDim.x + threadIdx.x;   // 2,097,152 total
  int half = (int)(i & 15);
  int lane = (int)((i >> 4) & 31);
  int nt   = (int)((i >> 9) & 15);
  int kt   = (int)((i >> 13) & 15);
  int j    = (int)((i >> 17) & 3);
  int l    = (int)(i >> 19);
  if (l >= NL) return;
  int k = kt * 32 + (half >> 3) * 16 + (lane >> 4) * 8 + (half & 7);
  int n = nt * 16 + (lane & 15);
  dst[i] = (_Float16)Wr[((long)(l * 4 + j) * HID + k) * HID + n];
}

__global__ void pack_wf(const float* __restrict__ Wf, _Float16* __restrict__ dst) {
  long i = (long)blockIdx.x * blockDim.x + threadIdx.x;   // 1,703,936 total
  const long TOT = (long)NL * 16 * NT_WF * 32 * 16;
  if (i >= TOT) return;
  int  half = (int)(i & 15);
  int  lane = (int)((i >> 4) & 31);
  long rest = i >> 9;
  int  nt   = (int)(rest % NT_WF);
  long r2   = rest / NT_WF;
  int  kt   = (int)(r2 & 15);
  int  l    = (int)(r2 >> 4);
  int k = kt * 32 + (half >> 3) * 16 + (lane >> 4) * 8 + (half & 7);
  int n = nt * 16 + (lane & 15);
  float v = (n < PTOT) ? Wf[((long)l * HID + k) * PTOT + n] : 0.0f;
  dst[i] = (_Float16)v;
}

// ---------------------------------------------------------------------------
// Fragment loaders
// ---------------------------------------------------------------------------
__device__ __forceinline__ v16h load_frag(const _Float16* p, int gap) {
  v8h lo = *(const v8h*)(p);
  v8h hi = *(const v8h*)(p + gap);
  return __builtin_shufflevector(lo, hi, 0,1,2,3,4,5,6,7,8,9,10,11,12,13,14,15);
}

#define WMMA(a, b, c) __builtin_amdgcn_wmma_f32_16x16x32_f16(false, (a), false, (b), \
                                                             (short)0, (c), false, false)

// 2x2 register tile: rows {0-15,16-31} x col tiles {nt0, nt1}. B fragments are
// reused across both row tiles, A fragments across both col tiles.
__device__ __forceinline__ void gemm_2x2(const _Float16* __restrict__ hA,
                                         const _Float16* __restrict__ Bpk,
                                         long blk0, long blk1, int blkStride, int lane,
                                         v8f& c00, v8f& c01, v8f& c10, v8f& c11) {
  const _Float16* arow0 = hA + (lane & 15) * HA_STRIDE + (lane >> 4) * 8;
  const _Float16* arow1 = arow0 + 16 * HA_STRIDE;
  const _Float16* bb0   = Bpk + blk0 * 512 + lane * 16;
  const _Float16* bb1   = Bpk + blk1 * 512 + lane * 16;
#pragma unroll
  for (int kt = 0; kt < 8; ++kt) {
    v16h a0 = load_frag(arow0 + kt * 32, 16);
    v16h a1 = load_frag(arow1 + kt * 32, 16);
    long bo = (long)kt * blkStride * 512;
    v16h b0 = load_frag(bb0 + bo, 8);
    v16h b1 = load_frag(bb1 + bo, 8);
    c00 = WMMA(a0, b0, c00);
    c01 = WMMA(a0, b1, c01);
    c10 = WMMA(a1, b0, c10);
    c11 = WMMA(a1, b1, c11);
  }
}

// 2x1 tile (both row tiles, one col tile) for the 26-tile Wf projection.
__device__ __forceinline__ void gemm_2x1(const _Float16* __restrict__ hA,
                                         const _Float16* __restrict__ Bpk,
                                         long blk0, int blkStride, int lane,
                                         v8f& cA, v8f& cB) {
  const _Float16* arow0 = hA + (lane & 15) * HA_STRIDE + (lane >> 4) * 8;
  const _Float16* arow1 = arow0 + 16 * HA_STRIDE;
  const _Float16* bb0   = Bpk + blk0 * 512 + lane * 16;
#pragma unroll
  for (int kt = 0; kt < 8; ++kt) {
    v16h a0 = load_frag(arow0 + kt * 32, 16);
    v16h a1 = load_frag(arow1 + kt * 32, 16);
    v16h b0 = load_frag(bb0 + (long)kt * blkStride * 512, 8);
    cA = WMMA(a0, b0, cA);
    cB = WMMA(a1, b0, cB);
  }
}

// D layout: lane: N = nt*16 + (lane&15); VGPR v: M = mtile*16 + v + 8*(lane>>4)
__device__ __forceinline__ void store_tile(float* __restrict__ dst, int dstride,
                                           int mtile, int nt, v8f acc,
                                           const float* __restrict__ bias,
                                           int nmax, int lane) {
  int N = nt * 16 + (lane & 15);
  if (N < nmax) {
    int Mb = mtile * 16 + (lane >> 4) * 8;
    float bvv = bias[N];
#pragma unroll
    for (int v = 0; v < 8; ++v) dst[(Mb + v) * dstride + N] = acc[v] + bvv;
  }
}

// Build f16 A operand from f32 source: relu(src (+ pe)) -> hA (row-major LDS)
__device__ __forceinline__ void buildA(const float* __restrict__ src, int sstride,
                                       bool use_pe, int t, _Float16* __restrict__ hA,
                                       const float* __restrict__ peL) {
  int rr = t >> 4;
  int cb = (t & 15) * 16;
#pragma unroll
  for (int p2 = 0; p2 < 2; ++p2) {
    int row = rr + 16 * p2;
#pragma unroll
    for (int i = 0; i < 16; ++i) {
      int c = cb + i;
      float v = src[row * sstride + c];
      if (use_pe) v += peL[c];
      hA[row * HA_STRIDE + c] = (_Float16)(v > 0.f ? v : 0.f);
    }
  }
}

// Rational-quadratic spline for one (row, feature) pair.
__device__ __forceinline__ void spline_eval(const float* __restrict__ P, float xv,
                                            float& yout, float& ldout) {
  float uw[NB], uh[NB], us[NB + 1];
#pragma unroll
  for (int i = 0; i < NB; ++i) uw[i] = P[i];
#pragma unroll
  for (int i = 0; i < NB; ++i) uh[i] = P[NB + i];
#pragma unroll
  for (int i = 0; i < NB + 1; ++i) us[i] = P[2 * NB + i];

  float mw = uw[0], mh = uh[0];
#pragma unroll
  for (int i = 1; i < NB; ++i) { mw = fmaxf(mw, uw[i]); mh = fmaxf(mh, uh[i]); }
  float sw = 0.f, shh = 0.f;
#pragma unroll
  for (int i = 0; i < NB; ++i) {
    uw[i] = __expf(uw[i] - mw); sw  += uw[i];
    uh[i] = __expf(uh[i] - mh); shh += uh[i];
  }
  const float span = 10.0f - NB * 1e-4f;
  float kw = span / sw, kh = span / shh;
  float xk[NB + 1], yk[NB + 1], dd[NB + 1];
  xk[0] = -5.f; yk[0] = -5.f;
#pragma unroll
  for (int i = 0; i < NB; ++i) {
    xk[i + 1] = xk[i] + (uw[i] * kw + 1e-4f);
    yk[i + 1] = yk[i] + (uh[i] * kh + 1e-4f);
  }
  const float OFFS = 0.5411655f;   // log(expm1(1 - MIN_SLOPE))
#pragma unroll
  for (int i = 0; i < NB + 1; ++i) {
    float tt = us[i] + OFFS;
    float sp = (tt > 15.f) ? tt : __logf(1.f + __expf(tt));
    dd[i] = sp + 1e-4f;
  }
  float x0 = xk[0], x1 = xk[1], y0 = yk[0], y1 = yk[1], d0 = dd[0], d1 = dd[1];
#pragma unroll
  for (int i = 1; i < NB; ++i) {
    bool c = xv >= xk[i];
    x0 = c ? xk[i]     : x0;
    x1 = c ? xk[i + 1] : x1;
    y0 = c ? yk[i]     : y0;
    y1 = c ? yk[i + 1] : y1;
    d0 = c ? dd[i]     : d0;
    d1 = c ? dd[i + 1] : d1;
  }
  float invw = 1.f / (x1 - x0);
  float s2   = (y1 - y0) * invw;
  float z    = fminf(fmaxf((xv - x0) * invw, 0.f), 1.f);
  float z1   = 1.f - z;
  float den  = s2 + (d1 + d0 - 2.f * s2) * z * z1;
  float yin  = y0 + (y1 - y0) * (s2 * z * z + d0 * z * z1) / den;
  float num  = d1 * z * z + 2.f * s2 * z * z1 + d0 * z1 * z1;
  float ldin = 2.f * __logf(s2) + __logf(num) - 2.f * __logf(den);
  bool inside = (xv >= -5.f) && (xv <= 5.f);
  yout  = inside ? yin : xv;
  ldout = inside ? ldin : 0.f;
}

// ---------------------------------------------------------------------------
// Fused flow kernel: 32 batch rows per block, 8 wave32 per block.
// ---------------------------------------------------------------------------
__global__ __launch_bounds__(256)
void flow_kernel(const float* __restrict__ xg,
                 const float* __restrict__ W0,
                 const float* __restrict__ b0g,
                 const float* __restrict__ brg,
                 const float* __restrict__ bfg,
                 const float* __restrict__ scales,
                 const float* __restrict__ shifts,
                 const int*   __restrict__ perms,
                 const _Float16* __restrict__ WrPk,
                 const _Float16* __restrict__ WfPk,
                 float* __restrict__ out) {
  __shared__ alignas(16) _Float16 hA[ROWS * HA_STRIDE];
  __shared__ float hF[ROWS * HID];
  __shared__ float tF[ROWS * T_STRIDE];
  __shared__ float xb[ROWS][NF];
  __shared__ float xm[ROWS][NF];
  __shared__ float red[ROWS][NF + 1];
  __shared__ float ldrow[ROWS];
  __shared__ float peL[HID];

  const int t    = threadIdx.x;
  const int lane = t & 31;
  const int w    = t >> 5;   // wave id 0..7
  const int rr   = t >> 4;   // base row 0..15; thread also owns row rr+16
  const int f    = t & 15;   // feature / column-group

  peL[t] = __sinf(6.283185307179586f * ((float)t / (float)HID));   // LOC_ALPHA = 1
  if (t < ROWS) ldrow[t] = 0.f;
  const long rowbase = (long)blockIdx.x * ROWS;
#pragma unroll
  for (int p2 = 0; p2 < 2; ++p2) {
    int r = rr + 16 * p2;
    xb[r][f] = xg[(rowbase + r) * NF + f];
  }
  __syncthreads();

#pragma unroll 1
  for (int l = 0; l < NL; ++l) {
    // ---- permutation + affine + mask -----------------------------------
    int   pidx  = perms[l * NF + f];
    float xold0 = xb[rr][pidx];
    float xold1 = xb[rr + 16][pidx];
    __syncthreads();
    float sc = scales[l * NF + f];
    float sh = shifts[l * NF + f];
    const bool trans = ((f & 1) == (l & 1));          // transformed this layer
    float xv0 = xold0 * sc + sh;
    float xv1 = xold1 * sc + sh;
    xb[rr][f]      = xv0;  xm[rr][f]      = trans ? 0.f : xv0;
    xb[rr + 16][f] = xv1;  xm[rr + 16][f] = trans ? 0.f : xv1;
    float ldc0 = __logf(fabsf(sc));
    float ldc1 = ldc0;
    __syncthreads();

    // ---- h = xm @ W0 + b0 (K=16; cheap VALU path, thread = column) -----
    {
      const int c = t;
      float bvv = b0g[l * HID + c];
#pragma unroll
      for (int half = 0; half < 2; ++half) {
        float acc[16];
#pragma unroll
        for (int q = 0; q < 16; ++q) acc[q] = 0.f;
#pragma unroll
        for (int ff = 0; ff < NF; ++ff) {
          float wv = W0[(l * NF + ff) * HID + c];
#pragma unroll
          for (int q = 0; q < 16; ++q) acc[q] += xm[half * 16 + q][ff] * wv;
        }
#pragma unroll
        for (int q = 0; q < 16; ++q) hF[(half * 16 + q) * HID + c] = acc[q] + bvv;
      }
    }
    __syncthreads();

    // ---- residual block 0 ----------------------------------------------
    buildA(hF, HID, true, t, hA, peL);                 // relu(h + pe)
    __syncthreads();
    {
      v8f c00 = {0,0,0,0,0,0,0,0}, c01 = c00, c10 = c00, c11 = c00;
      gemm_2x2(hA, WrPk, (long)(l * 4 + 0) * 256 + 2 * w,
                         (long)(l * 4 + 0) * 256 + 2 * w + 1, 16, lane,
               c00, c01, c10, c11);
      const float* bias = brg + (l * 4 + 0) * HID;
      store_tile(tF, T_STRIDE, 0, 2 * w,     c00, bias, HID, lane);
      store_tile(tF, T_STRIDE, 0, 2 * w + 1, c01, bias, HID, lane);
      store_tile(tF, T_STRIDE, 1, 2 * w,     c10, bias, HID, lane);
      store_tile(tF, T_STRIDE, 1, 2 * w + 1, c11, bias, HID, lane);
    }
    __syncthreads();
    buildA(tF, T_STRIDE, true, t, hA, peL);            // relu(t + pe)
    __syncthreads();
    {
      v8f c00 = {0,0,0,0,0,0,0,0}, c01 = c00, c10 = c00, c11 = c00;
      gemm_2x2(hA, WrPk, (long)(l * 4 + 1) * 256 + 2 * w,
                         (long)(l * 4 + 1) * 256 + 2 * w + 1, 16, lane,
               c00, c01, c10, c11);
      const float* bias = brg + (l * 4 + 1) * HID;
      store_tile(tF, T_STRIDE, 0, 2 * w,     c00, bias, HID, lane);
      store_tile(tF, T_STRIDE, 0, 2 * w + 1, c01, bias, HID, lane);
      store_tile(tF, T_STRIDE, 1, 2 * w,     c10, bias, HID, lane);
      store_tile(tF, T_STRIDE, 1, 2 * w + 1, c11, bias, HID, lane);
    }
    __syncthreads();
    {   // h += y ; A = relu(h + pe)  (fused, same element ownership)
      int cb = f * 16;
#pragma unroll
      for (int p2 = 0; p2 < 2; ++p2) {
        int row = rr + 16 * p2;
#pragma unroll
        for (int i = 0; i < 16; ++i) {
          int c = cb + i;
          float hv = hF[row * HID + c] + tF[row * T_STRIDE + c];
          hF[row * HID + c] = hv;
          float av = hv + peL[c];
          hA[row * HA_STRIDE + c] = (_Float16)(av > 0.f ? av : 0.f);
        }
      }
    }
    __syncthreads();

    // ---- residual block 1 ----------------------------------------------
    {
      v8f c00 = {0,0,0,0,0,0,0,0}, c01 = c00, c10 = c00, c11 = c00;
      gemm_2x2(hA, WrPk, (long)(l * 4 + 2) * 256 + 2 * w,
                         (long)(l * 4 + 2) * 256 + 2 * w + 1, 16, lane,
               c00, c01, c10, c11);
      const float* bias = brg + (l * 4 + 2) * HID;
      store_tile(tF, T_STRIDE, 0, 2 * w,     c00, bias, HID, lane);
      store_tile(tF, T_STRIDE, 0, 2 * w + 1, c01, bias, HID, lane);
      store_tile(tF, T_STRIDE, 1, 2 * w,     c10, bias, HID, lane);
      store_tile(tF, T_STRIDE, 1, 2 * w + 1, c11, bias, HID, lane);
    }
    __syncthreads();
    buildA(tF, T_STRIDE, true, t, hA, peL);
    __syncthreads();
    {
      v8f c00 = {0,0,0,0,0,0,0,0}, c01 = c00, c10 = c00, c11 = c00;
      gemm_2x2(hA, WrPk, (long)(l * 4 + 3) * 256 + 2 * w,
                         (long)(l * 4 + 3) * 256 + 2 * w + 1, 16, lane,
               c00, c01, c10, c11);
      const float* bias = brg + (l * 4 + 3) * HID;
      store_tile(tF, T_STRIDE, 0, 2 * w,     c00, bias, HID, lane);
      store_tile(tF, T_STRIDE, 0, 2 * w + 1, c01, bias, HID, lane);
      store_tile(tF, T_STRIDE, 1, 2 * w,     c10, bias, HID, lane);
      store_tile(tF, T_STRIDE, 1, 2 * w + 1, c11, bias, HID, lane);
    }
    __syncthreads();
    {   // h += y ; A = relu(h)  (no pe for final projection)
      int cb = f * 16;
#pragma unroll
      for (int p2 = 0; p2 < 2; ++p2) {
        int row = rr + 16 * p2;
#pragma unroll
        for (int i = 0; i < 16; ++i) {
          int c = cb + i;
          float hv = hF[row * HID + c] + tF[row * T_STRIDE + c];
          hF[row * HID + c] = hv;
          hA[row * HA_STRIDE + c] = (_Float16)(hv > 0.f ? hv : 0.f);
        }
      }
    }
    __syncthreads();

    // ---- p = relu(h) @ Wf + bf  (26 col tiles x 2 row tiles, wave-uniform)
#pragma unroll
    for (int s = 0; s < 4; ++s) {
      int nt = w + 8 * s;
      if (nt < NT_WF) {
        v8f cA = {0,0,0,0,0,0,0,0}, cB = cA;
        gemm_2x1(hA, WfPk, (long)l * (16 * NT_WF) + nt, NT_WF, lane, cA, cB);
        const float* bias = bfg + l * PTOT;
        store_tile(tF, T_STRIDE, 0, nt, cA, bias, PTOT, lane);
        store_tile(tF, T_STRIDE, 1, nt, cB, bias, PTOT, lane);
      }
    }
    __syncthreads();

    // ---- rational-quadratic spline, two (row, feature) pairs per thread -
    {
      float yv, ld;
      spline_eval(&tF[rr * T_STRIDE + f * PPF], xb[rr][f], yv, ld);
      if (trans) { xb[rr][f] = yv; ldc0 += ld; }
      red[rr][f] = ldc0;
    }
    {
      int r1 = rr + 16;
      float yv, ld;
      spline_eval(&tF[r1 * T_STRIDE + f * PPF], xb[r1][f], yv, ld);
      if (trans) { xb[r1][f] = yv; ldc1 += ld; }
      red[r1][f] = ldc1;
    }
    __syncthreads();
    if (t < ROWS) {
      float s3 = 0.f;
#pragma unroll
      for (int i = 0; i < NF; ++i) s3 += red[t][i];
      ldrow[t] += s3;
    }
    __syncthreads();
  }

  // ---- base log prob ----------------------------------------------------
#pragma unroll
  for (int p2 = 0; p2 < 2; ++p2) {
    int r = rr + 16 * p2;
    float xv3 = xb[r][f];
    red[r][f] = -0.5f * xv3 * xv3;
  }
  __syncthreads();
  if (t < ROWS) {
    float s4 = ldrow[t];
#pragma unroll
    for (int i = 0; i < NF; ++i) s4 += red[t][i];
    out[rowbase + t] = s4 - 14.703017f;   // 0.5 * 16 * log(2*pi)
  }
}

extern "C" void kernel_launch(void* const* d_in, const int* in_sizes, int n_in,
                              void* d_out, int out_size, void* d_ws, size_t ws_size,
                              hipStream_t stream) {
  const float* xg     = (const float*)d_in[0];
  const float* W0     = (const float*)d_in[1];
  const float* b0g    = (const float*)d_in[2];
  const float* Wr     = (const float*)d_in[3];
  const float* brg    = (const float*)d_in[4];
  const float* Wf     = (const float*)d_in[5];
  const float* bfg    = (const float*)d_in[6];
  const float* scales = (const float*)d_in[7];
  const float* shifts = (const float*)d_in[8];
  const int*   perms  = (const int*)d_in[9];
  float* out = (float*)d_out;

  _Float16* WrPk = (_Float16*)d_ws;                           // 8*4*256*256 halves = 4 MB
  _Float16* WfPk = WrPk + (size_t)NL * 4 * HID * HID;         // 8*16*26*512 halves

  pack_wr<<<8192, 256, 0, stream>>>(Wr, WrPk);                // 2,097,152 threads
  pack_wf<<<6656, 256, 0, stream>>>(Wf, WfPk);                // 1,703,936 threads

  int batch   = in_sizes[0] / NF;                             // 131072
  int nblocks = batch / ROWS;                                 // 4096
  flow_kernel<<<nblocks, 256, 0, stream>>>(xg, W0, b0g, brg, bfg, scales, shifts,
                                           perms, WrPk, WfPk, out);
}
